// SparseAttention_58042188038488
// MI455X (gfx1250) — matrix operand order repaired and checked by reading, not compiled, not run on hardware
//
#include <hip/hip_runtime.h>
#include <math.h>

typedef __attribute__((ext_vector_type(16))) _Float16 v16h;
typedef __attribute__((ext_vector_type(8)))  float    v8f;

namespace {
constexpr int BATCH = 4;
constexpr int HEADS = 8;
constexpr int NSEQ  = 4096;
constexpr int HD    = 64;
constexpr int BLK   = 32;
constexpr int NBLK  = 128;   // NSEQ / BLK
constexpr int MM    = 8;     // max kept key blocks per block row
constexpr float SCALE = 0.125f;  // 64^-0.5

union Frag {
  v16h v;
  _Float16 h[16];
  unsigned u[8];
  uint4 q4[2];
};

__device__ __forceinline__ void cvt4(const float4 f, _Float16* d) {
  d[0] = (_Float16)f.x; d[1] = (_Float16)f.y;
  d[2] = (_Float16)f.z; d[3] = (_Float16)f.w;
}
} // namespace

__global__ __launch_bounds__(256) void sparse_attn_wmma(
    const float* __restrict__ qg, const float* __restrict__ kg,
    const float* __restrict__ vg, const int* __restrict__ cols,
    float* __restrict__ outg)
{
  __shared__ __align__(16) _Float16 Pbuf[BLK][MM * BLK];   // 16 KB probs
  __shared__ float wmax[8][16];
  __shared__ float wsum[8][16];
  __shared__ int colsr[MM];
  __shared__ int validr[MM];

  const int bid = blockIdx.x;
  const int r  = bid % NBLK;        // block row
  const int bh = bid / NBLK;        // fused batch*head

  const int tid  = threadIdx.x;
  const int wave = tid >> 5;
  const int lane = tid & 31;
  const int half = lane >> 4;
  const int l16  = lane & 15;

  if (tid < MM) colsr[tid] = cols[r * MM + tid];
  __syncthreads();
  if (tid == 0) {
    validr[0] = 1;  // diagonal block always present
    for (int j = 1; j < MM; ++j) validr[j] = (colsr[j] > colsr[j - 1]) ? 1 : 0;
  }
  __syncthreads();

  const size_t base = (size_t)bh * NSEQ * HD;
  const float* Q = qg + base;
  const float* K = kg + base;
  const float* V = vg + base;

  // ---------------- Phase 1: S^T = K' @ Q^T (each lane owns ONE query) -----
  const int qt      = wave & 1;          // query half (rows 0-15 / 16-31)
  const int ktbase  = (wave >> 1) * 4;   // this wave's 4 key tiles (of 16)
  const int qrow_in = qt * 16 + l16;     // query row within block (0..31)

  // B operand = Q^T for this wave's query tile; two K=32 chunks.
  // Layout: h[0..7] = Q[qrow][chunk*32 + 8*half + {0..7}], h[8..15] = Q[qrow][chunk*32 + 16 + 8*half + {0..7}]
  Frag bq0, bq1;
  {
    const float* qp = Q + (size_t)(r * BLK + qrow_in) * HD + 8 * half;
    cvt4(*(const float4*)(qp +  0), &bq0.h[0]);
    cvt4(*(const float4*)(qp +  4), &bq0.h[4]);
    cvt4(*(const float4*)(qp + 16), &bq0.h[8]);
    cvt4(*(const float4*)(qp + 20), &bq0.h[12]);
    cvt4(*(const float4*)(qp + 32), &bq1.h[0]);
    cvt4(*(const float4*)(qp + 36), &bq1.h[4]);
    cvt4(*(const float4*)(qp + 48), &bq1.h[8]);
    cvt4(*(const float4*)(qp + 52), &bq1.h[12]);
  }

  float sc[4][8];      // per-lane scores (32 keys for one query)
  int aliveMask = 0;   // scalar: which of this wave's 4 tiles are valid
#pragma unroll
  for (int tt = 0; tt < 4; ++tt) {
    const int kt = ktbase + tt;
    // key-block index is uniform per 16-key tile: (kt*16 + 0..15) >> 5 == kt>>1
    const int mb = kt >> 1;
    const int sval = __builtin_amdgcn_readfirstlane(validr[mb]);
    if (!sval) continue;              // scalar branch: skip loads + WMMA
    aliveMask |= (1 << tt);
    const int scol = __builtin_amdgcn_readfirstlane(colsr[mb]);
    const int cin  = (kt * 16 + l16) & 31;     // key-in-block for A rows
    // A layout: h[0..7] = K[row][chunk*32 + 8*half + {0..7}],
    //           h[8..15] = K[row][chunk*32 + 16 + 8*half + {0..7}]
    const float* kp = K + (size_t)(scol * BLK + cin) * HD + 8 * half;
    Frag a0, a1;
    cvt4(*(const float4*)(kp +  0), &a0.h[0]);
    cvt4(*(const float4*)(kp +  4), &a0.h[4]);
    cvt4(*(const float4*)(kp + 16), &a0.h[8]);
    cvt4(*(const float4*)(kp + 20), &a0.h[12]);
    cvt4(*(const float4*)(kp + 32), &a1.h[0]);
    cvt4(*(const float4*)(kp + 36), &a1.h[4]);
    cvt4(*(const float4*)(kp + 48), &a1.h[8]);
    cvt4(*(const float4*)(kp + 52), &a1.h[12]);
    v8f c = {};
    c = __builtin_amdgcn_wmma_f32_16x16x32_f16(false, a0.v, false, bq0.v,
                                               (short)0, c, false, false);
    c = __builtin_amdgcn_wmma_f32_16x16x32_f16(false, a1.v, false, bq1.v,
                                               (short)0, c, false, false);
    // single scale path (packed f32 muls), no duplicated branch arms
#pragma unroll
    for (int i = 0; i < 8; ++i) sc[tt][i] = c[i] * SCALE;
    if (scol == r) {
      // diagonal block: causal tril overwrite (scalar-guarded, 2 tiles/WG)
      const int cibase = (kt * 16 + 8 * half) & 31;  // i<8 cannot carry past 31
#pragma unroll
      for (int i = 0; i < 8; ++i)
        if (cibase + i > qrow_in) sc[tt][i] = -INFINITY;
    }
  }

  // ---------------- Softmax over kept keys of this lane's query ------------
  float rowmax = -INFINITY;
#pragma unroll
  for (int tt = 0; tt < 4; ++tt)
    if (aliveMask & (1 << tt))
#pragma unroll
      for (int i = 0; i < 8; ++i) rowmax = fmaxf(rowmax, sc[tt][i]);
  rowmax = fmaxf(rowmax, __shfl_xor(rowmax, 16, 32));
  if (lane < 16) wmax[wave][l16] = rowmax;
  __syncthreads();
  float gm = -INFINITY;
#pragma unroll
  for (int w2 = 0; w2 < 4; ++w2) gm = fmaxf(gm, wmax[2 * w2 + qt][l16]);

  float psum = 0.f;
#pragma unroll
  for (int tt = 0; tt < 4; ++tt)
    if (aliveMask & (1 << tt))
#pragma unroll
      for (int i = 0; i < 8; ++i) {
        float e = __expf(sc[tt][i] - gm);   // -inf -> 0
        sc[tt][i] = e;
        psum += e;
      }
  psum += __shfl_xor(psum, 16, 32);
  if (lane < 16) wsum[wave][l16] = psum;
  __syncthreads();
  float tot = 0.f;
#pragma unroll
  for (int w2 = 0; w2 < 4; ++w2) tot += wsum[2 * w2 + qt][l16];
  const float inv = 1.0f / tot;

  // Write normalized probs row-major into LDS: Pbuf[query][key position].
  // Dead tiles are skipped; phase 2 never reads those columns.
#pragma unroll
  for (int tt = 0; tt < 4; ++tt)
    if (aliveMask & (1 << tt)) {
      union { _Float16 h[8]; uint4 u; } pk;
#pragma unroll
      for (int i = 0; i < 8; ++i) pk.h[i] = (_Float16)(sc[tt][i] * inv);
      *(uint4*)&Pbuf[qrow_in][(ktbase + tt) * 16 + 8 * half] = pk.u;
    }
  __syncthreads();

  // ---------------- Phase 2: O = P @ V' (one 16x16 tile per wave) ----------
  const int rt = wave >> 2;           // output row tile (0/1)
  const int dt = wave & 3;            // output col tile (0..3)
  const int dcol = dt * 16 + l16;
  v8f acc = {};
#pragma unroll
  for (int mb = 0; mb < MM; ++mb) {
    const int sval = __builtin_amdgcn_readfirstlane(validr[mb]);
    if (!sval) continue;              // scalar branch: EXEC untouched
    const int scol = __builtin_amdgcn_readfirstlane(colsr[mb]);
    Frag a, bv;
    // A fragments: two contiguous 16B LDS reads per chunk
    a.q4[0] = *(const uint4*)&Pbuf[rt * 16 + l16][mb * BLK + 8 * half];
    a.q4[1] = *(const uint4*)&Pbuf[rt * 16 + l16][mb * BLK + 16 + 8 * half];
    const float* vp = V + (size_t)(scol * BLK + 8 * half) * HD + dcol;
#pragma unroll
    for (int j = 0; j < 8; ++j) {
      bv.h[j]     = (_Float16)vp[(size_t)(j) * HD];
      bv.h[j + 8] = (_Float16)vp[(size_t)(j + 16) * HD];
    }
    acc = __builtin_amdgcn_wmma_f32_16x16x32_f16(false, a.v, false, bv.v,
                                                 (short)0, acc, false, false);
  }

  float* O = outg + base + (size_t)(r * BLK) * HD;
#pragma unroll
  for (int i = 0; i < 8; ++i)
    O[(size_t)(rt * 16 + 8 * half + i) * HD + dcol] = acc[i];
}

extern "C" void kernel_launch(void* const* d_in, const int* in_sizes, int n_in,
                              void* d_out, int out_size, void* d_ws, size_t ws_size,
                              hipStream_t stream) {
  (void)in_sizes; (void)n_in; (void)d_ws; (void)ws_size; (void)out_size;
  const float* q   = (const float*)d_in[0];
  const float* k   = (const float*)d_in[1];
  const float* v   = (const float*)d_in[2];
  const int* cols  = (const int*)d_in[3];
  // d_in[4] (valid) is intentionally unused: validity is recomputed from the
  // strictly-ascending structure of cols (padding breaks monotonicity).
  float* out = (float*)d_out;

  dim3 grid(BATCH * HEADS * NBLK);   // 4096 workgroups
  dim3 block(256);                   // 8 wave32 waves
  sparse_attn_wmma<<<grid, block, 0, stream>>>(q, k, v, cols, out);
}
